// ParticleNetDisCo_50087908606116
// MI455X (gfx1250) — compile-verified, matrix operand order checked by reading, not run on hardware
//
#include <hip/hip_runtime.h>
#include <hip/hip_bf16.h>

typedef __attribute__((ext_vector_type(16))) _Float16 v16h;
typedef __attribute__((ext_vector_type(8)))  _Float16 v8h;
typedef __attribute__((ext_vector_type(8)))  float    v8f;

constexpr int B  = 64;
constexpr int P  = 1024;
constexpr int KN = 7;      // neighbors kept
constexpr float COORD_SHIFT = 1000000000.0f;

// ---------------------------------------------------------------------------
// WMMA fragment loader.
// 16-bit A matrix 16x32 layout (ISA 7.12.2):
//   lanes 0-15 : M=lane,   halves 0..7 = K 0..7,  halves 8..15 = K 16..23
//   lanes 16-31: M=lane-16,halves 0..7 = K 8..15, halves 8..15 = K 24..31
// B matrix (K x N) needs column n over K with the same per-lane K pattern,
// i.e. row n of the row-major (N x K) weight array -> same loader works for
// both A tiles (row-major activations) and B tiles (row-major (out,in) weights).
// ---------------------------------------------------------------------------
__device__ __forceinline__ v16h load_frag(const _Float16* base, int ldk, int lane) {
  int m  = lane & 15;
  int h8 = (lane >> 4) << 3;           // 0 or 8
  const _Float16* row = base + m * ldk + h8;
  v8h lo = *(const v8h*)(row);         // K = h8 .. h8+7
  v8h hi = *(const v8h*)(row + 16);    // K = 16+h8 .. 16+h8+7
  v16h r;
#pragma unroll
  for (int j = 0; j < 8; ++j) { r[j] = lo[j]; r[8 + j] = hi[j]; }
  return r;
}

// One 1x1-conv layer: rows 0..127 (8 waves x 16-row M tiles), relu((x@W^T)*g+b)
__device__ __forceinline__ void conv_layer(const _Float16* in, int kdim,
                                           const _Float16* wgt,
                                           const float* __restrict__ g,
                                           const float* __restrict__ bb,
                                           _Float16* out, int cout,
                                           int wave, int lane) {
  int n0    = lane & 15;
  int half8 = (lane >> 4) << 3;
  for (int nt = 0; nt < cout / 16; ++nt) {
    v8f acc = {};
    for (int kc = 0; kc < kdim / 32; ++kc) {
      v16h a  = load_frag(in  + wave * 16 * kdim + kc * 32, kdim, lane);
      v16h bf = load_frag(wgt + nt   * 16 * kdim + kc * 32, kdim, lane);
      acc = __builtin_amdgcn_wmma_f32_16x16x32_f16(false, a, false, bf,
                                                   (short)0, acc, false, false);
    }
    int   n  = nt * 16 + n0;
    float gg = g[n], bv = bb[n];
#pragma unroll
    for (int r = 0; r < 8; ++r) {
      float v = acc[r] * gg + bv;
      v = v > 0.f ? v : 0.f;
      out[(wave * 16 + half8 + r) * cout + n] = (_Float16)v;   // D: M=r+8*half
    }
  }
}

// ---------------------------------------------------------------------------
// 1) preprocess: masked coords, folded BN on features
// ---------------------------------------------------------------------------
__global__ void preprocess_kernel(const float* __restrict__ pts,
                                  const float* __restrict__ feat,
                                  const float* __restrict__ mask,
                                  const float* __restrict__ bng,
                                  const float* __restrict__ bnb,
                                  float* __restrict__ coords1,
                                  float* __restrict__ fts0) {
  int i = blockIdx.x * blockDim.x + threadIdx.x;
  if (i >= B * P) return;
  int b = i >> 10, p = i & (P - 1);
  float m = mask[i];
#pragma unroll
  for (int c = 0; c < 2; ++c)
    coords1[(b * 2 + c) * P + p] = pts[(b * 2 + c) * P + p] * m;
#pragma unroll
  for (int c = 0; c < 16; ++c) {
    float v = feat[(b * 16 + c) * P + p] * m;
    fts0[(b * 16 + c) * P + p] = (v * bng[c] + bnb[c]) * m;
  }
}

__global__ void zero_kernel(float* __restrict__ ptr, int n) {
  int i = blockIdx.x * blockDim.x + threadIdx.x;
  if (i < n) ptr[i] = 0.f;
}

// ---------------------------------------------------------------------------
// 2) kNN: top-8 of d = 2*inner - xx_p - xx_q (largest d = nearest), drop self.
//    coords are already masked; +COORD_SHIFT applied here for masked points.
//    grid = B*4 blocks x 256 threads; LDS tiles of 256 candidate points.
// ---------------------------------------------------------------------------
template <int C>
__global__ __launch_bounds__(256) void knn_kernel(const float* __restrict__ coords,
                                                  const float* __restrict__ mask,
                                                  int* __restrict__ idx_out) {
  __shared__ float tile[C * 256];
  __shared__ float txx[256];
  int tid = threadIdx.x;
  int b   = blockIdx.x >> 2;
  int p   = ((blockIdx.x & 3) << 8) + tid;

  float sp = (mask[b * P + p] == 0.f) ? COORD_SHIFT : 0.f;
  float my[C];
  float xxp = 0.f;
#pragma unroll
  for (int c = 0; c < C; ++c) {
    my[c] = coords[(b * C + c) * P + p] + sp;
    xxp += my[c] * my[c];
  }

  float bd[8]; int bi[8];
#pragma unroll
  for (int j = 0; j < 8; ++j) { bd[j] = -3.0e38f; bi[j] = 0x7FFFFFFF; }

  for (int t = 0; t < 4; ++t) {
    int q0 = t * 256;
    float sq = (mask[b * P + q0 + tid] == 0.f) ? COORD_SHIFT : 0.f;
    float a = 0.f;
#pragma unroll
    for (int c = 0; c < C; ++c) {
      float v = coords[(b * C + c) * P + q0 + tid] + sq;
      tile[c * 256 + tid] = v;
      a += v * v;
    }
    txx[tid] = a;
    __syncthreads();
    for (int j = 0; j < 256; ++j) {
      int q = q0 + j;
      float inner = 0.f;
#pragma unroll
      for (int c = 0; c < C; ++c) inner += my[c] * tile[c * 256 + j];
      float d = 2.f * inner - xxp - txx[j];
      // insert with tie-break on smaller index (JAX top_k semantics)
      if (d > bd[7] || (d == bd[7] && q < bi[7])) {
        bd[7] = d; bi[7] = q;
#pragma unroll
        for (int s = 7; s > 0; --s) {
          bool up = bd[s] > bd[s - 1] || (bd[s] == bd[s - 1] && bi[s] < bi[s - 1]);
          if (up) {
            float td = bd[s]; bd[s] = bd[s - 1]; bd[s - 1] = td;
            int   ti = bi[s]; bi[s] = bi[s - 1]; bi[s - 1] = ti;
          }
        }
      }
    }
    __syncthreads();
  }
#pragma unroll
  for (int j = 0; j < KN; ++j)
    idx_out[(b * P + p) * KN + j] = bi[j + 1];   // drop self (rank 0)
}

// ---------------------------------------------------------------------------
// 3) fused EdgeConv block: gather -> [ctr, nbr-ctr] -> 3 WMMA conv layers ->
//    mean over k -> shortcut -> relu -> *mask.
//    One block = one batch x 16 points (112 real rows, padded to 128).
// ---------------------------------------------------------------------------
template <int CIN, int COUT>
__global__ __launch_bounds__(256) void edgeconv_kernel(
    const float* __restrict__ fts_in, const int* __restrict__ idx,
    const float* __restrict__ mask,
    const float* __restrict__ W0, const float* __restrict__ W1,
    const float* __restrict__ W2,
    const float* __restrict__ g0, const float* __restrict__ g1,
    const float* __restrict__ g2,
    const float* __restrict__ b0, const float* __restrict__ b1,
    const float* __restrict__ b2,
    const float* __restrict__ sc_w, const float* __restrict__ sc_g,
    const float* __restrict__ sc_b,
    float* __restrict__ fts_out) {
  constexpr int K2   = 2 * CIN;
  constexpr int ABUF = (K2 > COUT ? K2 : COUT);
  __shared__ alignas(16) _Float16 w0h[COUT * K2];
  __shared__ alignas(16) _Float16 w1h[COUT * COUT];
  __shared__ alignas(16) _Float16 w2h[COUT * COUT];
  __shared__ alignas(16) _Float16 bufA[128 * ABUF];
  __shared__ alignas(16) _Float16 bufB[128 * COUT];
  __shared__ float ctr[16 * CIN];

  int tid = threadIdx.x, wave = tid >> 5, lane = tid & 31;
  int b   = blockIdx.x >> 6;
  int pt0 = (blockIdx.x & 63) << 4;

  for (int i = tid; i < COUT * K2; i += 256) w0h[i] = (_Float16)W0[i];
  for (int i = tid; i < COUT * COUT; i += 256) {
    w1h[i] = (_Float16)W1[i];
    w2h[i] = (_Float16)W2[i];
  }
  for (int i = tid; i < 16 * CIN; i += 256) {
    int pp = i / CIN, c = i % CIN;
    ctr[i] = fts_in[(b * CIN + c) * P + pt0 + pp];
  }
  __syncthreads();

  // build x = [ctr, nbr - ctr] in f16 (rows 0..111; 112..127 are pad)
  for (int i = tid; i < 112 * CIN; i += 256) {
    int row = i / CIN, c = i % CIN;
    int pp = row / KN, kk = row - pp * KN;
    int q  = idx[(b * P + pt0 + pp) * KN + kk];
    float fc = ctr[pp * CIN + c];
    float nb = fts_in[(b * CIN + c) * P + q];
    bufA[row * K2 + c]       = (_Float16)fc;
    bufA[row * K2 + CIN + c] = (_Float16)(nb - fc);
  }
  __syncthreads();

  conv_layer(bufA, K2,   w0h, g0, b0, bufB, COUT, wave, lane);
  __syncthreads();
  conv_layer(bufB, COUT, w1h, g1, b1, bufA, COUT, wave, lane);
  __syncthreads();
  conv_layer(bufA, COUT, w2h, g2, b2, bufB, COUT, wave, lane);
  __syncthreads();

  // mean over k + shortcut + relu + mask
  for (int i = tid; i < 16 * COUT; i += 256) {
    int pp = i / COUT, c = i % COUT;
    float s = 0.f;
#pragma unroll
    for (int kk = 0; kk < KN; ++kk) s += (float)bufB[(pp * KN + kk) * COUT + c];
    float mean = s * (1.f / 7.f);
    float sc = 0.f;
    for (int ci = 0; ci < CIN; ++ci) sc += ctr[pp * CIN + ci] * sc_w[c * CIN + ci];
    float o = sc * sc_g[c] + sc_b[c] + mean;
    o = o > 0.f ? o : 0.f;
    o *= mask[b * P + pt0 + pp];
    fts_out[(b * COUT + c) * P + pt0 + pp] = o;
  }
}

// ---------------------------------------------------------------------------
// 4) fusion conv (96->128) + masked sum over P into pooled[B][128]
//    One block = one batch x 128 points; 8 waves x 16-row M tiles.
// ---------------------------------------------------------------------------
__global__ __launch_bounds__(256) void fusion_kernel(
    const float* __restrict__ fts1, const float* __restrict__ fts2,
    const float* __restrict__ mask,
    const float* __restrict__ fus_w, const float* __restrict__ fus_g,
    const float* __restrict__ fus_b,
    float* __restrict__ pooled) {
  __shared__ alignas(16) _Float16 xk[128 * 96];
  __shared__ alignas(16) _Float16 wfh[128 * 96];
  __shared__ float msk[128];
  __shared__ float fsum[128];
  int tid = threadIdx.x, wave = tid >> 5, lane = tid & 31;
  int b  = blockIdx.x >> 3;
  int p0 = (blockIdx.x & 7) << 7;

  for (int i = tid; i < 128 * 96; i += 256) wfh[i] = (_Float16)fus_w[i];
  for (int i = tid; i < 128 * 96; i += 256) {
    int row = i / 96, c = i - (i / 96) * 96;
    int p = p0 + row;
    float v = (c < 32) ? fts1[(b * 32 + c) * P + p]
                       : fts2[(b * 64 + (c - 32)) * P + p];
    xk[row * 96 + c] = (_Float16)v;
  }
  if (tid < 128) { msk[tid] = mask[b * P + p0 + tid]; fsum[tid] = 0.f; }
  __syncthreads();

  int n0 = lane & 15, half8 = (lane >> 4) << 3;
  for (int nt = 0; nt < 8; ++nt) {
    v8f acc = {};
    for (int kc = 0; kc < 3; ++kc) {
      v16h a  = load_frag(xk  + wave * 16 * 96 + kc * 32, 96, lane);
      v16h bf = load_frag(wfh + nt   * 16 * 96 + kc * 32, 96, lane);
      acc = __builtin_amdgcn_wmma_f32_16x16x32_f16(false, a, false, bf,
                                                   (short)0, acc, false, false);
    }
    int   n  = nt * 16 + n0;
    float gg = fus_g[n], bb = fus_b[n];
    float part = 0.f;
#pragma unroll
    for (int r = 0; r < 8; ++r) {
      float v = acc[r] * gg + bb;
      v = v > 0.f ? v : 0.f;
      part += v * msk[wave * 16 + half8 + r];
    }
    atomicAdd(&fsum[n], part);          // ds_add_f32
  }
  __syncthreads();
  if (tid < 128) atomicAdd(&pooled[b * 128 + tid], fsum[tid]);  // global f32 atomic
}

// ---------------------------------------------------------------------------
// 5) head: counts, pooled mean, fc1 relu, fc2
// ---------------------------------------------------------------------------
__global__ __launch_bounds__(128) void head_kernel(
    const float* __restrict__ pooled, const float* __restrict__ mask,
    const float* __restrict__ fc1_w, const float* __restrict__ fc1_b,
    const float* __restrict__ fc2_w, const float* __restrict__ fc2_b,
    float* __restrict__ out) {
  __shared__ float red[128];
  __shared__ float pool[128];
  __shared__ float h1[128];
  int b = blockIdx.x, t = threadIdx.x;
  float cs = 0.f;
  for (int p = t; p < P; p += 128) cs += mask[b * P + p];
  red[t] = cs;
  __syncthreads();
  for (int s = 64; s > 0; s >>= 1) {
    if (t < s) red[t] += red[t + s];
    __syncthreads();
  }
  float count = red[0] > 1.f ? red[0] : 1.f;
  pool[t] = pooled[b * 128 + t] / count;
  __syncthreads();
  float a = fc1_b[t];
  for (int k = 0; k < 128; ++k) a += pool[k] * fc1_w[t * 128 + k];
  h1[t] = a > 0.f ? a : 0.f;
  __syncthreads();
  if (t < 2) {
    float o = fc2_b[t];
    for (int k = 0; k < 128; ++k) o += h1[k] * fc2_w[t * 128 + k];
    out[b * 2 + t] = o;
  }
}

// ---------------------------------------------------------------------------
extern "C" void kernel_launch(void* const* d_in, const int* in_sizes, int n_in,
                              void* d_out, int out_size, void* d_ws, size_t ws_size,
                              hipStream_t stream) {
  (void)in_sizes; (void)n_in; (void)out_size; (void)ws_size;
  const float* points   = (const float*)d_in[0];
  const float* features = (const float*)d_in[1];
  const float* mask     = (const float*)d_in[2];
  // JAX pytree flatten order (dict keys sorted; 'Ws' < 'bs' < 'gs' < 'sc_*')
  int i = 3;
  const float* bn_b = (const float*)d_in[i++];   // bn_fts_b
  const float* bn_g = (const float*)d_in[i++];   // bn_fts_g
  const float* e0W0 = (const float*)d_in[i++];
  const float* e0W1 = (const float*)d_in[i++];
  const float* e0W2 = (const float*)d_in[i++];
  const float* e0b0 = (const float*)d_in[i++];
  const float* e0b1 = (const float*)d_in[i++];
  const float* e0b2 = (const float*)d_in[i++];
  const float* e0g0 = (const float*)d_in[i++];
  const float* e0g1 = (const float*)d_in[i++];
  const float* e0g2 = (const float*)d_in[i++];
  const float* e0scb = (const float*)d_in[i++];
  const float* e0scg = (const float*)d_in[i++];
  const float* e0scw = (const float*)d_in[i++];
  const float* e1W0 = (const float*)d_in[i++];
  const float* e1W1 = (const float*)d_in[i++];
  const float* e1W2 = (const float*)d_in[i++];
  const float* e1b0 = (const float*)d_in[i++];
  const float* e1b1 = (const float*)d_in[i++];
  const float* e1b2 = (const float*)d_in[i++];
  const float* e1g0 = (const float*)d_in[i++];
  const float* e1g1 = (const float*)d_in[i++];
  const float* e1g2 = (const float*)d_in[i++];
  const float* e1scb = (const float*)d_in[i++];
  const float* e1scg = (const float*)d_in[i++];
  const float* e1scw = (const float*)d_in[i++];
  const float* fc1_b = (const float*)d_in[i++];
  const float* fc1_w = (const float*)d_in[i++];
  const float* fc2_b = (const float*)d_in[i++];
  const float* fc2_w = (const float*)d_in[i++];
  const float* fus_b = (const float*)d_in[i++];
  const float* fus_g = (const float*)d_in[i++];
  const float* fus_w = (const float*)d_in[i++];

  // workspace carve-up
  float* fts0    = (float*)d_ws;                       // B*16*P
  float* coords1 = fts0 + B * 16 * P;                  // B*2*P
  int*   idx1    = (int*)(coords1 + B * 2 * P);        // B*P*7
  float* fts1    = (float*)(idx1 + B * P * KN);        // B*32*P
  int*   idx2    = (int*)(fts1 + B * 32 * P);          // B*P*7
  float* fts2    = (float*)(idx2 + B * P * KN);        // B*64*P
  float* pooled  = fts2 + B * 64 * P;                  // B*128

  preprocess_kernel<<<(B * P) / 256, 256, 0, stream>>>(points, features, mask,
                                                       bn_g, bn_b, coords1, fts0);
  zero_kernel<<<(B * 128 + 255) / 256, 256, 0, stream>>>(pooled, B * 128);

  knn_kernel<2><<<B * 4, 256, 0, stream>>>(coords1, mask, idx1);

  edgeconv_kernel<16, 32><<<B * (P / 16), 256, 0, stream>>>(
      fts0, idx1, mask, e0W0, e0W1, e0W2, e0g0, e0g1, e0g2,
      e0b0, e0b1, e0b2, e0scw, e0scg, e0scb, fts1);

  knn_kernel<32><<<B * 4, 256, 0, stream>>>(fts1, mask, idx2);

  edgeconv_kernel<32, 64><<<B * (P / 16), 256, 0, stream>>>(
      fts1, idx2, mask, e1W0, e1W1, e1W2, e1g0, e1g1, e1g2,
      e1b0, e1b1, e1b2, e1scw, e1scg, e1scb, fts2);

  fusion_kernel<<<B * (P / 128), 256, 0, stream>>>(fts1, fts2, mask,
                                                   fus_w, fus_g, fus_b, pooled);

  head_kernel<<<B, 128, 0, stream>>>(pooled, mask, fc1_w, fc1_b, fc2_w, fc2_b,
                                     (float*)d_out);
}
